// C2LMGAgent_69252052680781
// MI455X (gfx1250) — matrix operand order repaired and checked by reading, not compiled, not run on hardware
//
#include <hip/hip_runtime.h>
#include <math.h>
#include <stdint.h>

// Problem constants
#define BZ 64
#define NA 64
#define DM 128
#define RR 3
#define NH 4
#define INF_ 96
#define NACT 20
#define NROW (BZ*NA)          // 4096 agent rows
#define PST 136               // padded LDS row stride (ushorts) to break bank conflicts

typedef __attribute__((ext_vector_type(16))) __bf16 v16bf;
typedef __attribute__((ext_vector_type(8)))  float  v8f;

// ---------- helpers ----------
__device__ __forceinline__ unsigned short f2bf(float f) {
    unsigned u = __float_as_uint(f);
    unsigned r = u + 0x7fffu + ((u >> 16) & 1u);   // round-to-nearest-even
    return (unsigned short)(r >> 16);
}
__device__ __forceinline__ float bf2f(unsigned short u) {
    return __uint_as_float(((unsigned)u) << 16);
}
__device__ __forceinline__ float sigf(float x) { return 1.f / (1.f + __expf(-x)); }

// 16 contiguous bf16 (A and B fragments here: lane holds K-run of 16, k = 16h + s)
__device__ __forceinline__ v16bf ld_frag_contig(const unsigned short* p) {
    union { uint4 q[2]; v16bf v; } u;
    u.q[0] = *(const uint4*)(p);
    u.q[1] = *(const uint4*)(p + 8);
    return u.v;
}
// A-matrix fragment: two 8-element chunks at p and p+16 (ISA 16-bit A 16x32 layout)
__device__ __forceinline__ v16bf ld_frag_split(const unsigned short* p) {
    union { uint4 q[2]; v16bf v; } u;
    u.q[0] = *(const uint4*)(p);
    u.q[1] = *(const uint4*)(p + 16);
    return u.v;
}

// CDNA5 async LDS<->global copies (ASYNCcnt-tracked); inline asm is portable
// across ROCm 7.2 and amdgpu-toolchain per the bridge doc.
__device__ __forceinline__ void async_load_b128(unsigned lds_off, const void* gaddr) {
    asm volatile("global_load_async_to_lds_b128 %0, %1, off"
                 :: "v"(lds_off), "v"((unsigned long long)(uintptr_t)gaddr) : "memory");
}
__device__ __forceinline__ void async_store_b128(const void* gaddr, unsigned lds_off) {
    asm volatile("global_store_async_from_lds_b128 %0, %1, off"
                 :: "v"((unsigned long long)(uintptr_t)gaddr), "v"(lds_off) : "memory");
}
__device__ __forceinline__ void wait_async0() {
    asm volatile("s_wait_asynccnt 0x0" ::: "memory");
}

// ============================================================
// K0: weight conversion to bf16 (W_m2 full, W_s1 message half)
// ============================================================
__global__ __launch_bounds__(256) void k_cvt(const float* __restrict__ W_m2,
                                             const float* __restrict__ W_s1,
                                             unsigned short* __restrict__ wm2bf,
                                             unsigned short* __restrict__ ws1bf) {
    int i = blockIdx.x * 256 + threadIdx.x;
    if (i < DM * DM) wm2bf[i] = f2bf(W_m2[i]);
    if (i < RR * 64 * DM) {
        int r = i / (64 * DM);
        int o = (i / DM) % 64;
        int d = i % DM;
        ws1bf[i] = f2bf(W_s1[(r * 64 + o) * 256 + 128 + d]);  // message half
    }
}

// ============================================================
// K1: encoder + GRU + others + A/B + q + hs   (one block per agent row)
// ============================================================
__global__ __launch_bounds__(128) void k_agent(
    const float* __restrict__ obs, const float* __restrict__ hid,
    const float* __restrict__ W_enc, const float* __restrict__ b_enc,
    const float* __restrict__ W_ih,  const float* __restrict__ W_hh,
    const float* __restrict__ b_ih,  const float* __restrict__ b_hh,
    const float* __restrict__ W_oth, const float* __restrict__ b_oth,
    const float* __restrict__ W_m1,  const float* __restrict__ b_m1,
    const float* __restrict__ W_s1,  const float* __restrict__ b_s1,
    const float* __restrict__ Wq,
    float* __restrict__ g_h, float* __restrict__ g_A, float* __restrict__ g_B,
    float* __restrict__ g_q, float* __restrict__ g_hs) {
    int row = blockIdx.x;
    int t = threadIdx.x;
    __shared__ float s_obs[INF_], s_hid[DM], s_x[DM], s_gx[3*DM], s_gh[3*DM], s_h[DM], s_oth[DM];
    if (t < INF_) s_obs[t] = obs[row * INF_ + t];
    if (t < DM)   s_hid[t] = hid[row * DM + t];
    __syncthreads();
    {   // x = obs @ W_enc.T + b_enc
        float acc = b_enc[t];
        const float* w = W_enc + t * INF_;
        for (int c = 0; c < INF_; ++c) acc += s_obs[c] * w[c];
        s_x[t] = acc;
    }
    __syncthreads();
    for (int s = 0; s < 3; ++s) {   // GRU gate pre-activations
        int o = t + 128 * s;
        float ax = b_ih[o], ah = b_hh[o];
        const float* wx = W_ih + o * DM;
        const float* wh = W_hh + o * DM;
        for (int d = 0; d < DM; ++d) { ax += s_x[d] * wx[d]; ah += s_hid[d] * wh[d]; }
        s_gx[o] = ax; s_gh[o] = ah;
    }
    __syncthreads();
    {   // GRU combine (r,z,n order)
        float r = sigf(s_gx[t] + s_gh[t]);
        float z = sigf(s_gx[128 + t] + s_gh[128 + t]);
        float n = tanhf(s_gx[256 + t] + r * s_gh[256 + t]);
        float h = (1.f - z) * n + z * s_hid[t];
        s_h[t] = h;
        g_h[row * DM + t] = h;
    }
    __syncthreads();
    {   // others = h @ W_oth.T + b_oth
        float acc = b_oth[t];
        const float* w = W_oth + t * DM;
        for (int d = 0; d < DM; ++d) acc += s_h[d] * w[d];
        s_oth[t] = acc;
    }
    __syncthreads();
    {   // A = others @ W_m1[:, :D].T ; B = h @ W_m1[:, D:].T + b_m1
        float a = 0.f, bb = b_m1[t];
        const float* wa = W_m1 + t * 256;
        const float* wb = wa + 128;
        for (int d = 0; d < DM; ++d) { a += s_oth[d] * wa[d]; bb += s_h[d] * wb[d]; }
        g_A[row * DM + t] = a;
        g_B[row * DM + t] = bb;
    }
    for (int r = 0; r < RR; ++r) {  // q = h @ Wq[r].T
        float acc = 0.f;
        const float* w = Wq + ((size_t)r * DM + t) * DM;
        for (int d = 0; d < DM; ++d) acc += s_h[d] * w[d];
        g_q[((size_t)r * NROW + row) * DM + t] = acc;
    }
    if (t < 64) {                   // hs = h @ W_s1[r][:, :D].T + b_s1
        for (int r = 0; r < RR; ++r) {
            float acc = b_s1[r * 64 + t];
            const float* w = W_s1 + ((size_t)r * 64 + t) * 256;
            for (int d = 0; d < DM; ++d) acc += s_h[d] * w[d];
            g_hs[((size_t)r * NROW + row) * 64 + t] = acc;
        }
    }
}

// ============================================================
// K2: WMMA bf16 — message layer-2 GEMM + sched-diff GEMM
//     one block per (b,i); 4 waves; each wave = one 16-row j tile.
//     Message tile writeback = async LDS->HBM, overlapped with sched WMMAs.
// ============================================================
__global__ __launch_bounds__(128) void k_msg(
    const float* __restrict__ g_A, const float* __restrict__ g_B,
    const float* __restrict__ b_m2,
    const unsigned short* __restrict__ wm2bf, const unsigned short* __restrict__ ws1bf,
    const float* __restrict__ g_hs,
    const float* __restrict__ W_s2, const float* __restrict__ b_s2,
    unsigned short* __restrict__ g_msg, float* __restrict__ g_diff) {
    __shared__ unsigned short s_pre[64 * PST];
    __shared__ unsigned short s_msg[64 * PST];
    int blk = blockIdx.x;
    int b = blk >> 6, i = blk & 63;
    int t = threadIdx.x, lane = t & 31, wv = t >> 5;
    int lm = lane & 15, lh = (lane >> 4) & 1;

    // stage 1: pre = relu(A[b,j] + B[b,i]) -> bf16 in LDS
    const float* Brow = g_B + (size_t)blk * DM;
    for (int idx = t; idx < 64 * DM; idx += 128) {
        int j = idx >> 7, d = idx & 127;
        float v = g_A[(((size_t)b * 64 + j) << 7) + d] + Brow[d];
        s_pre[j * PST + d] = f2bf(fmaxf(v, 0.f));
    }
    __syncthreads();

    // stage 2: message[j,n] = relu(pre @ W_m2.T + b_m2) -> bf16 in LDS
    int jt = wv;
    const unsigned short* prow = s_pre + (jt * 16 + lm) * PST;
    v16bf afr[4];
#pragma unroll
    for (int kt = 0; kt < 4; ++kt) afr[kt] = ld_frag_contig(prow + kt * 32 + lh * 16);
#pragma unroll
    for (int nt = 0; nt < 8; ++nt) {
        v8f acc = {};
        int n = nt * 16 + lm;
#pragma unroll
        for (int kt = 0; kt < 4; ++kt) {
            v16bf bfr = ld_frag_contig(wm2bf + n * DM + kt * 32 + lh * 16);
            acc = __builtin_amdgcn_wmma_f32_16x16x32_bf16(false, afr[kt], false, bfr,
                                                          (short)0, acc, false, false);
        }
        float bias = b_m2[n];
#pragma unroll
        for (int v = 0; v < 8; ++v) {
            int m = jt * 16 + v + lh * 8;
            s_msg[m * PST + n] = f2bf(fmaxf(acc[v] + bias, 0.f));
        }
    }
    __syncthreads();

    // async writeback of the whole message tile (1024 x 16B), overlapped with stage 3
    for (int c = t; c < 1024; c += 128) {
        int j = c >> 4, ch = c & 15;
        unsigned lds = (unsigned)(uintptr_t)(s_msg + j * PST + ch * 8);
        const void* ga = (const void*)(g_msg + (size_t)blk * (64 * DM) + j * DM + ch * 8);
        async_store_b128(ga, lds);
    }

    // stage 3: sched logit-diff per round (M=64,K=128,N=64 WMMA + shfl reduce)
    const unsigned short* mrow = s_msg + (jt * 16 + lm) * PST;
    v16bf amf[4];
#pragma unroll
    for (int kt = 0; kt < 4; ++kt) amf[kt] = ld_frag_contig(mrow + kt * 32 + lh * 16);
#pragma unroll
    for (int r = 0; r < RR; ++r) {
        float srow[8];
#pragma unroll
        for (int v = 0; v < 8; ++v) srow[v] = 0.f;
        const float* hsrow = g_hs + ((size_t)r * NROW + blk) * 64;
#pragma unroll
        for (int nt = 0; nt < 4; ++nt) {
            v8f acc = {};
            int n = nt * 16 + lm;
#pragma unroll
            for (int kt = 0; kt < 4; ++kt) {
                v16bf bfr = ld_frag_contig(ws1bf + (((size_t)r * 64 + n) << 7) + kt * 32 + lh * 16);
                acc = __builtin_amdgcn_wmma_f32_16x16x32_bf16(false, amf[kt], false, bfr,
                                                              (short)0, acc, false, false);
            }
            float hb  = hsrow[n];
            float w2d = W_s2[r * 128 + 64 + n] - W_s2[r * 128 + n];
#pragma unroll
            for (int v = 0; v < 8; ++v) {
                float u = fmaxf(acc[v] + hb, 0.f);
                srow[v] += u * w2d;
            }
        }
#pragma unroll
        for (int v = 0; v < 8; ++v) {   // reduce over n across the 16 lanes of each half
            float s = srow[v];
            s += __shfl_xor(s, 1); s += __shfl_xor(s, 2);
            s += __shfl_xor(s, 4); s += __shfl_xor(s, 8);
            srow[v] = s;
        }
        if (lm == 0) {
            float bdel = b_s2[r * 2 + 1] - b_s2[r * 2 + 0];
#pragma unroll
            for (int v = 0; v < 8; ++v) {
                int m = jt * 16 + v + lh * 8;
                g_diff[((size_t)r * NROW + blk) * 64 + m] = srow[v] + bdel;
            }
        }
    }
    wait_async0();   // drain async stores before wave exit
}

// ============================================================
// K3: fused 3-round masked attention; one block per (b,i)
//     Message tile loaded via async global->LDS copies.
// ============================================================
__global__ __launch_bounds__(128) void k_attn(
    const unsigned short* __restrict__ g_msg, const float* __restrict__ g_q,
    const float* __restrict__ g_diff, const float* __restrict__ gumbel,
    const float* __restrict__ Wk, const float* __restrict__ Wv, const float* __restrict__ Wo,
    float* __restrict__ g_aggre) {
    __shared__ unsigned short s_msg[64 * PST];
    __shared__ float s_qk[NH * DM], s_sc[64 * NH], s_w[64 * NH], s_adj[64];
    __shared__ float s_mh[NH * DM], s_att[DM], s_agg[DM];
    int blk = blockIdx.x;
    int b = blk >> 6, i = blk & 63;
    int t = threadIdx.x;
    // async load of the message tile: 1024 x 16B directly into LDS
    for (int c = t; c < 1024; c += 128) {
        int j = c >> 4, ch = c & 15;
        unsigned lds = (unsigned)(uintptr_t)(s_msg + j * PST + ch * 8);
        const void* ga = (const void*)(g_msg + (size_t)blk * (64 * DM) + j * DM + ch * 8);
        async_load_b128(lds, ga);
    }
    s_agg[t] = 0.f;
    wait_async0();
    __syncthreads();
    const float scale = 0.17677669529663687f;  // 1/sqrt(32)
    for (int r = 0; r < RR; ++r) {
        // qk[h][d] = sum_e q[h*32+e] * Wk[r][h*32+e][d]  (k-projection collapsed)
        const float* qrow = g_q + ((size_t)r * NROW + blk) * DM;
        for (int idx = t; idx < NH * DM; idx += 128) {
            int h = idx >> 7, d = idx & 127;
            float acc = 0.f;
            const float* wk = Wk + (size_t)r * DM * DM + (h * 32) * DM + d;
            for (int e = 0; e < 32; ++e) acc += qrow[h * 32 + e] * wk[e * DM];
            s_qk[idx] = acc;
        }
        __syncthreads();
        // scores[j][h] = scale * msg[j] . qk[h]
        for (int idx = t; idx < 64 * NH; idx += 128) {
            int j = idx >> 2, h = idx & 3;
            const unsigned short* mr = s_msg + j * PST;
            const float* qk = s_qk + h * DM;
            float acc = 0.f;
            for (int d = 0; d < DM; ++d) acc += bf2f(mr[d]) * qk[d];
            s_sc[j * NH + h] = acc * scale;
        }
        // adj (straight-through gumbel argmax of 2 classes)
        if (t < 64) {
            int j = t;
            float dij = g_diff[((size_t)r * NROW + blk) * 64 + j];
            float dji = g_diff[((size_t)r * NROW + b * 64 + j) * 64 + i];
            const float* g = gumbel + ((((((size_t)r * BZ + b) * NA + i) * NA + j)) << 1);
            float delta = 0.5f * (dij + dji) + (g[1] - g[0]);
            s_adj[j] = (delta > 0.f) ? 1.f : 0.f;
        }
        __syncthreads();
        // masked, renormalized softmax over j per head
        if (t < NH) {
            int h = t;
            float mx = -1e30f;
            for (int j = 0; j < 64; ++j) mx = fmaxf(mx, s_sc[j * NH + h]);
            float sum = 0.f;
            for (int j = 0; j < 64; ++j) {
                float e = __expf(s_sc[j * NH + h] - mx);
                s_w[j * NH + h] = e; sum += e;
            }
            float sum2 = 0.f;
            for (int j = 0; j < 64; ++j) {
                float wm = (s_w[j * NH + h] / sum) * s_adj[j];
                s_w[j * NH + h] = wm; sum2 += wm;
            }
            float inv = 1.f / (sum2 + 1e-10f);
            for (int j = 0; j < 64; ++j) s_w[j * NH + h] *= inv;
        }
        __syncthreads();
        // mh[h][d] = sum_j w[j,h] * msg[j,d]   (v-projection collapsed)
        for (int idx = t; idx < NH * DM; idx += 128) {
            int h = idx >> 7, d = idx & 127;
            float acc = 0.f;
            for (int j = 0; j < 64; ++j) acc += s_w[j * NH + h] * bf2f(s_msg[j * PST + d]);
            s_mh[idx] = acc;
        }
        __syncthreads();
        // att[d] = mh[h(d)] . Wv[r][d]
        {
            int h = t >> 5;
            float acc = 0.f;
            const float* wv = Wv + (size_t)r * DM * DM + t * DM;
            const float* mh = s_mh + h * DM;
            for (int d = 0; d < DM; ++d) acc += mh[d] * wv[d];
            s_att[t] = acc;
        }
        __syncthreads();
        // out[n] = elu(att @ Wo.T); accumulate mean
        {
            float acc = 0.f;
            const float* wo = Wo + (size_t)r * DM * DM + t * DM;
            for (int d = 0; d < DM; ++d) acc += s_att[d] * wo[d];
            float e = (acc > 0.f) ? acc : (__expf(acc) - 1.f);
            s_agg[t] += e;
        }
        __syncthreads();
    }
    g_aggre[(size_t)blk * DM + t] = s_agg[t] * (1.f / 3.f);
}

// ============================================================
// K4: q-head + h passthrough
// ============================================================
__global__ __launch_bounds__(128) void k_final(
    const float* __restrict__ g_h, const float* __restrict__ g_aggre,
    const float* __restrict__ W_q1, const float* __restrict__ b_q1,
    const float* __restrict__ W_q2, const float* __restrict__ b_q2,
    float* __restrict__ d_out) {
    int row = blockIdx.x, t = threadIdx.x;
    __shared__ float s_in[2 * DM], s_t1[DM];
    s_in[t] = g_h[(size_t)row * DM + t];
    s_in[DM + t] = g_aggre[(size_t)row * DM + t];
    __syncthreads();
    {
        float acc = b_q1[t];
        const float* w = W_q1 + t * 2 * DM;
        for (int c = 0; c < 2 * DM; ++c) acc += s_in[c] * w[c];
        s_t1[t] = fmaxf(acc, 0.f);
    }
    __syncthreads();
    if (t < NACT) {
        float acc = b_q2[t];
        const float* w = W_q2 + t * DM;
        for (int c = 0; c < DM; ++c) acc += s_t1[c] * w[c];
        d_out[(size_t)row * NACT + t] = acc;
    }
    d_out[(size_t)NROW * NACT + (size_t)row * DM + t] = s_in[t];  // h output
}

// ============================================================
extern "C" void kernel_launch(void* const* d_in, const int* in_sizes, int n_in,
                              void* d_out, int out_size, void* d_ws, size_t ws_size,
                              hipStream_t stream) {
    (void)in_sizes; (void)n_in; (void)out_size; (void)ws_size;
    const float* obs    = (const float*)d_in[0];
    const float* hidden = (const float*)d_in[1];
    const float* gumbel = (const float*)d_in[2];
    const float* W_enc  = (const float*)d_in[3];
    const float* b_enc  = (const float*)d_in[4];
    const float* W_ih   = (const float*)d_in[5];
    const float* W_hh   = (const float*)d_in[6];
    const float* b_ih   = (const float*)d_in[7];
    const float* b_hh   = (const float*)d_in[8];
    const float* W_oth  = (const float*)d_in[9];
    const float* b_oth  = (const float*)d_in[10];
    const float* W_m1   = (const float*)d_in[11];
    const float* b_m1   = (const float*)d_in[12];
    const float* W_m2   = (const float*)d_in[13];
    const float* b_m2   = (const float*)d_in[14];
    const float* W_s1   = (const float*)d_in[15];
    const float* b_s1   = (const float*)d_in[16];
    const float* W_s2   = (const float*)d_in[17];
    const float* b_s2   = (const float*)d_in[18];
    const float* Wq     = (const float*)d_in[19];
    const float* Wk     = (const float*)d_in[20];
    const float* Wv     = (const float*)d_in[21];
    const float* Wo     = (const float*)d_in[22];
    const float* W_q1   = (const float*)d_in[23];
    const float* b_q1   = (const float*)d_in[24];
    const float* W_q2   = (const float*)d_in[25];
    const float* b_q2   = (const float*)d_in[26];
    float* out = (float*)d_out;

    // workspace layout (bytes)
    char* ws = (char*)d_ws;
    size_t off = 0;
    float* g_h     = (float*)(ws + off); off += (size_t)NROW * DM * 4;        // 2 MB
    float* g_A     = (float*)(ws + off); off += (size_t)NROW * DM * 4;
    float* g_B     = (float*)(ws + off); off += (size_t)NROW * DM * 4;
    float* g_q     = (float*)(ws + off); off += (size_t)RR * NROW * DM * 4;   // 6 MB
    float* g_hs    = (float*)(ws + off); off += (size_t)RR * NROW * 64 * 4;   // 3 MB
    float* g_diff  = (float*)(ws + off); off += (size_t)RR * NROW * 64 * 4;   // 3 MB
    float* g_aggre = (float*)(ws + off); off += (size_t)NROW * DM * 4;
    unsigned short* g_msg = (unsigned short*)(ws + off); off += (size_t)NROW * 64 * DM * 2; // 67 MB
    unsigned short* wm2bf = (unsigned short*)(ws + off); off += (size_t)DM * DM * 2;
    unsigned short* ws1bf = (unsigned short*)(ws + off); off += (size_t)RR * 64 * DM * 2;

    k_cvt<<<dim3((RR * 64 * DM + 255) / 256), dim3(256), 0, stream>>>(W_m2, W_s1, wm2bf, ws1bf);
    k_agent<<<dim3(NROW), dim3(128), 0, stream>>>(obs, hidden, W_enc, b_enc, W_ih, W_hh,
                                                  b_ih, b_hh, W_oth, b_oth, W_m1, b_m1,
                                                  W_s1, b_s1, Wq, g_h, g_A, g_B, g_q, g_hs);
    k_msg<<<dim3(NROW), dim3(128), 0, stream>>>(g_A, g_B, b_m2, wm2bf, ws1bf, g_hs,
                                                W_s2, b_s2, g_msg, g_diff);
    k_attn<<<dim3(NROW), dim3(128), 0, stream>>>(g_msg, g_q, g_diff, gumbel, Wk, Wv, Wo, g_aggre);
    k_final<<<dim3(NROW), dim3(128), 0, stream>>>(g_h, g_aggre, W_q1, b_q1, W_q2, b_q2, out);
}
